// AutoModelRetrain_12275016532007
// MI455X (gfx1250) — matrix operand order repaired and checked by reading, not compiled, not run on hardware
//
#include <hip/hip_runtime.h>
#include <math.h>

#define EMB     16
#define MAXLEN  50
#define BATCH   32768
#define BN_EPS  1e-5f

typedef __attribute__((ext_vector_type(16))) _Float16 v16h;
typedef __attribute__((ext_vector_type(8)))  float    v8f;

// ---------------------------------------------------------------------------
// helpers
// ---------------------------------------------------------------------------

// sum across the 16 lanes of a half-wave (lane groups 0-15 / 16-31)
__device__ __forceinline__ float hred16(float v) {
    v += __shfl_xor(v, 1, 32);
    v += __shfl_xor(v, 2, 32);
    v += __shfl_xor(v, 4, 32);
    v += __shfl_xor(v, 8, 32);
    return v;
}

// A fragment (16x32 f16), ISA 7.12.2: lane holds row M=lane%16, kb=(lane/16)*8;
// VGPR j<4 -> K = kb+2j, kb+2j+1 ; VGPR j>=4 -> K = kb+16+...
__device__ __forceinline__ v16h load_a_frag(const _Float16* base, int ldk, int k0) {
    int lane = threadIdx.x & 31;
    int row  = lane & 15;
    int kb   = (lane >> 4) << 3;
    const _Float16* p = base + row * ldk + k0 + kb;
    v16h a;
#pragma unroll
    for (int j = 0; j < 4; ++j) {
        a[2 * j]         = p[2 * j];
        a[2 * j + 1]     = p[2 * j + 1];
        a[8 + 2 * j]     = p[16 + 2 * j];
        a[8 + 2 * j + 1] = p[16 + 2 * j + 1];
    }
    return a;
}

// B fragment (32x16 f16): lane holds column N=lane%16, kb=(lane/16)*8;
// element i<8 -> K = kb+i ; element i>=8 -> K = kb+16+(i-8).
__device__ __forceinline__ v16h load_b_frag(const _Float16* base, int ldn, int k0, int n0) {
    int lane = threadIdx.x & 31;
    const _Float16* p = base + (k0 + ((lane >> 4) << 3)) * ldn + n0 + (lane & 15);
    v16h b;
#pragma unroll
    for (int i = 0; i < 16; ++i) b[i] = p[((i & 7) + ((i >> 3) << 4)) * ldn];
    return b;
}

// ---------------------------------------------------------------------------
// kernel 0: zero the BN accumulators
// ---------------------------------------------------------------------------
__global__ void zero_kernel(float* __restrict__ ws) {
    int i = threadIdx.x;
    if (i < 128) ws[i] = 0.f;
}

// ---------------------------------------------------------------------------
// kernel 1: gather + 4 poolings + BN partial sums
// 256 threads: 16 lanes per row (one lane per EMB dim), 16 rows per block
// ---------------------------------------------------------------------------
__global__ __launch_bounds__(256) void pool_kernel(
    const int* __restrict__ ug,  const int* __restrict__ urb,
    const int* __restrict__ mg,  const int* __restrict__ mt,
    const int* __restrict__ ugm, const int* __restrict__ urbm,
    const int* __restrict__ mgm, const int* __restrict__ mtm,
    const float* __restrict__ embMovie, const float* __restrict__ embTag,
    const float* __restrict__ embGenre,
    const float* __restrict__ atMovie,  const float* __restrict__ atTag,
    const float* __restrict__ atGenre,
    float* __restrict__ gsum, float* __restrict__ gsq,
    float* __restrict__ pooled)
{
    __shared__ float pbuf[16][64];

    int tid  = threadIdx.x;
    int lane = tid & 31;
    int warp = tid >> 5;
    int rl   = warp * 2 + (lane >> 4);   // 0..15 local row
    int d    = lane & 15;                // embedding dim
    int b    = blockIdx.x * 16 + rl;

    // ---- feature 0: user_genre, 'sum' ----
    {
        int cnt = ugm[b];
        const int* idp = ug + b * MAXLEN;
        float acc = 0.f;
        for (int j = 0; j < cnt; ++j)
            acc += embGenre[idp[j] * EMB + d];
        pbuf[rl][0 * 16 + d] = acc;
        pooled[(size_t)b * 64 + 0 * 16 + d] = acc;
    }

    // ---- feature 1: urb, 'atten' (online softmax over all 50 incl. masked 0s) ----
    {
        int cnt = urbm[b];
        const int* idp = urb + b * MAXLEN;
        float m = -INFINITY, s = 0.f, acc = 0.f;
        for (int j = 0; j < cnt; ++j) {
            int id  = idp[j];
            float w = atMovie[id];
            float e = embMovie[id * EMB + d];
            float mn   = fmaxf(m, w);
            float corr = __expf(m - mn);
            float p    = __expf(w - mn);
            s   = s * corr + p;
            acc = acc * corr + p * e;
            m = mn;
        }
        // masked tail: (50-cnt) entries with weight 0, embedding 0
        {
            float mn   = fmaxf(m, 0.f);
            float corr = __expf(m - mn);
            float p    = __expf(0.f - mn) * (float)(MAXLEN - cnt);
            s   = s * corr + p;
            acc = acc * corr;
        }
        float r = acc / s;
        pbuf[rl][1 * 16 + d] = r;
        pooled[(size_t)b * 64 + 1 * 16 + d] = r;
    }

    // ---- feature 2: movie_genre, 'max' (argmax of L2 over L, first occurrence) ----
    {
        int cnt = mgm[b];
        const int* idp = mg + b * MAXLEN;
        float best = -1.f, keep = 0.f;
        for (int j = 0; j < cnt; ++j) {
            float e  = embGenre[idp[j] * EMB + d];
            float l2 = hred16(e * e);
            if (l2 > best) { best = l2; keep = e; }
        }
        pbuf[rl][2 * 16 + d] = keep;
        pooled[(size_t)b * 64 + 2 * 16 + d] = keep;
    }

    // ---- feature 3: movie_tag, 'korder' ----
    {
        int cnt = mtm[b];
        const int* idp = mt + b * MAXLEN;
        float s = 0.f, ss = 0.f;
        for (int j = 0; j < cnt; ++j) {
            float e = embTag[idp[j] * EMB + d];
            s  += e;
            ss += e * e;
        }
        float v   = 0.5f * (s * s - ss);
        float nsq = hred16(v * v);
        float n   = fmaxf(sqrtf(nsq), 1e-12f);
        float r   = v / n;
        pbuf[rl][3 * 16 + d] = r;
        pooled[(size_t)b * 64 + 3 * 16 + d] = r;
    }

    __syncthreads();

    // block-level BN partial sums -> global atomics (64 channels)
    if (tid < 64) {
        float s = 0.f, q = 0.f;
#pragma unroll
        for (int r = 0; r < 16; ++r) {
            float v = pbuf[r][tid];
            s += v; q += v * v;
        }
        atomicAdd(&gsum[tid], s);
        atomicAdd(&gsq[tid], q);
    }
}

// ---------------------------------------------------------------------------
// kernel 2: finalize BN -> per-channel scale/shift
// ---------------------------------------------------------------------------
__global__ void stats_kernel(const float* __restrict__ gsum, const float* __restrict__ gsq,
                             const float* __restrict__ gamma, const float* __restrict__ beta,
                             float* __restrict__ scale, float* __restrict__ shift)
{
    int c = threadIdx.x;
    if (c < 64) {
        const float inv = 1.f / (float)BATCH;
        float mu  = gsum[c] * inv;
        float var = gsq[c] * inv - mu * mu;
        float sc  = gamma[c] * rsqrtf(var + BN_EPS);
        scale[c] = sc;
        shift[c] = beta[c] - mu * sc;
    }
}

// ---------------------------------------------------------------------------
// kernel 3: assemble x = [emb1 | BN(pooled)] (K padded to 128) and run the MLP
// with WMMA. 128 threads = 4 waves; each wave owns a 16-row stripe.
// ---------------------------------------------------------------------------
__global__ __launch_bounds__(128) void mlp_kernel(
    const int* __restrict__ uid, const int* __restrict__ mid, const int* __restrict__ yr,
    const float* __restrict__ embU, const float* __restrict__ embM, const float* __restrict__ embY,
    const float* __restrict__ scale, const float* __restrict__ shift,
    const float* __restrict__ pooled,
    const float* __restrict__ W1, const float* __restrict__ b1,
    const float* __restrict__ W2, const float* __restrict__ b2,
    const float* __restrict__ W3, const float* __restrict__ b3,
    float* __restrict__ out)
{
    __shared__ _Float16 W1s[128][64];   // K x N (rows 112..127 zero)
    __shared__ _Float16 W2s[64][32];
    __shared__ _Float16 xt[4][16][128]; // per-wave 16 x 128 input tile
    __shared__ _Float16 h1[4][16][64];
    __shared__ float    h2[4][16][32];
    __shared__ float    b1s[64], b2s[32], W3s[32];

    int tid  = threadIdx.x;
    int lane = tid & 31;
    int w    = tid >> 5;

    // stage weights (block-wide)
    for (int i = tid; i < 128 * 64; i += 128) {
        int r = i >> 6, c = i & 63;
        W1s[r][c] = (_Float16)(r < 112 ? W1[r * 64 + c] : 0.f);
    }
    for (int i = tid; i < 64 * 32; i += 128)
        W2s[i >> 5][i & 31] = (_Float16)W2[i];
    if (tid < 64) b1s[tid] = b1[tid];
    if (tid < 32) { b2s[tid] = b2[tid]; W3s[tid] = W3[tid]; }

    // fill this wave's x tile: cols [0,48)=emb1 gathers, [48,112)=BN(pooled), [112,128)=0
    int r0 = blockIdx.x * 64 + w * 16;
    for (int i = lane; i < 16 * 128; i += 32) {
        int r = i >> 7, c = i & 127;
        int b = r0 + r;
        float v;
        if (c < 16)       v = embU[uid[b] * EMB + c];
        else if (c < 32)  v = embM[mid[b] * EMB + (c - 16)];
        else if (c < 48)  v = embY[yr[b]  * EMB + (c - 32)];
        else if (c < 112) { int k = c - 48; v = pooled[(size_t)b * 64 + k] * scale[k] + shift[k]; }
        else              v = 0.f;
        xt[w][r][c] = (_Float16)v;
    }
    __syncthreads();

    int crow = (lane >> 4) * 8;  // C/D: row = vgpr + 8*(lane/16)
    int ccol = lane & 15;        //      col = lane%16

    // ---- layer 1: [16x128] x [128x64] ----
#pragma unroll
    for (int nt = 0; nt < 4; ++nt) {
        v8f acc;
        float bb = b1s[nt * 16 + ccol];
#pragma unroll
        for (int r = 0; r < 8; ++r) acc[r] = bb;
#pragma unroll
        for (int ks = 0; ks < 4; ++ks) {
            v16h a  = load_a_frag(&xt[w][0][0], 128, ks * 32);
            v16h bf = load_b_frag(&W1s[0][0], 64, ks * 32, nt * 16);
            acc = __builtin_amdgcn_wmma_f32_16x16x32_f16(false, a, false, bf,
                                                         (short)0, acc, false, false);
        }
#pragma unroll
        for (int r = 0; r < 8; ++r)
            h1[w][crow + r][nt * 16 + ccol] = (_Float16)fmaxf(acc[r], 0.f);
    }
    __syncthreads();

    // ---- layer 2: [16x64] x [64x32] ----
#pragma unroll
    for (int nt = 0; nt < 2; ++nt) {
        v8f acc;
        float bb = b2s[nt * 16 + ccol];
#pragma unroll
        for (int r = 0; r < 8; ++r) acc[r] = bb;
#pragma unroll
        for (int ks = 0; ks < 2; ++ks) {
            v16h a  = load_a_frag(&h1[w][0][0], 64, ks * 32);
            v16h bf = load_b_frag(&W2s[0][0], 32, ks * 32, nt * 16);
            acc = __builtin_amdgcn_wmma_f32_16x16x32_f16(false, a, false, bf,
                                                         (short)0, acc, false, false);
        }
#pragma unroll
        for (int r = 0; r < 8; ++r)
            h2[w][crow + r][nt * 16 + ccol] = fmaxf(acc[r], 0.f);
    }
    __syncthreads();

    // ---- layer 3: [16x32] . [32] + sigmoid ----
    if (lane < 16) {
        float acc = b3[0];
#pragma unroll
        for (int k = 0; k < 32; ++k) acc += h2[w][lane][k] * W3s[k];
        out[r0 + lane] = 1.f / (1.f + __expf(-acc));
    }
}

// ---------------------------------------------------------------------------
// launch
// ---------------------------------------------------------------------------
extern "C" void kernel_launch(void* const* d_in, const int* in_sizes, int n_in,
                              void* d_out, int out_size, void* d_ws, size_t ws_size,
                              hipStream_t stream) {
    const int*   uid  = (const int*)d_in[0];
    const int*   mid  = (const int*)d_in[1];
    const int*   yr   = (const int*)d_in[2];
    const int*   ug   = (const int*)d_in[3];
    const int*   urb  = (const int*)d_in[4];
    const int*   mg   = (const int*)d_in[5];
    const int*   mt   = (const int*)d_in[6];
    const int*   ugm  = (const int*)d_in[7];
    const int*   urbm = (const int*)d_in[8];
    const int*   mgm  = (const int*)d_in[9];
    const int*   mtm  = (const int*)d_in[10];
    const float* embU = (const float*)d_in[11];
    const float* embM = (const float*)d_in[12];
    const float* embT = (const float*)d_in[13];
    const float* embG = (const float*)d_in[14];
    const float* embY = (const float*)d_in[15];
    const float* atM  = (const float*)d_in[16];
    const float* atT  = (const float*)d_in[17];
    const float* atG  = (const float*)d_in[18];
    const float* gam  = (const float*)d_in[19];
    const float* bet  = (const float*)d_in[20];
    const float* W1   = (const float*)d_in[21];
    const float* b1   = (const float*)d_in[22];
    const float* W2   = (const float*)d_in[23];
    const float* b2   = (const float*)d_in[24];
    const float* W3   = (const float*)d_in[25];
    const float* b3   = (const float*)d_in[26];

    float* ws     = (float*)d_ws;
    float* gsum   = ws;          // 64
    float* gsq    = ws + 64;     // 64
    float* scale  = ws + 128;    // 64
    float* shift  = ws + 192;    // 64
    float* pooled = ws + 256;    // B*64

    hipLaunchKernelGGL(zero_kernel, dim3(1), dim3(128), 0, stream, ws);

    hipLaunchKernelGGL(pool_kernel, dim3(BATCH / 16), dim3(256), 0, stream,
                       ug, urb, mg, mt, ugm, urbm, mgm, mtm,
                       embM, embT, embG, atM, atT, atG,
                       gsum, gsq, pooled);

    hipLaunchKernelGGL(stats_kernel, dim3(1), dim3(64), 0, stream,
                       gsum, gsq, gam, bet, scale, shift);

    hipLaunchKernelGGL(mlp_kernel, dim3(BATCH / 64), dim3(128), 0, stream,
                       uid, mid, yr, embU, embM, embY,
                       scale, shift, pooled,
                       W1, b1, W2, b2, W3, b3,
                       (float*)d_out);
}
